// MyLSTM_25142738550993
// MI455X (gfx1250) — matrix operand order repaired and checked by reading, not compiled, not run on hardware
//
#include <hip/hip_runtime.h>
#include <hip/hip_bf16.h>

// ---------------- types ----------------
typedef __bf16 bf16_t;
typedef __attribute__((ext_vector_type(16))) __bf16 v16bf;
typedef __attribute__((ext_vector_type(8)))  __bf16 v8bf;
typedef __attribute__((ext_vector_type(8)))  float  v8f;

#define B_    256
#define T_    128
#define F_    256
#define H0_   1024
#define H1_   256
#define OUTL_ 12

// ---------------- WMMA fragment loaders (bf16, 16x16x32) ----------------
// A-matrix 16x32 (MxK): lane(0..15)=row M, lanes 16..31 same rows with K-offset +16;
// each lane holds K = kh..kh+15, kh=(lane>>4)*16 (matches B fragment k order).
__device__ __forceinline__ v16bf load_a_frag(const bf16_t* __restrict__ base,
                                             long row_stride, int m0, int k0, int lane) {
    const int row = m0 + (lane & 15);
    const int kh  = (lane >> 4) * 16;
    const bf16_t* p = base + (long)row * row_stride + k0 + kh;
    v8bf lo = *(const v8bf*)(p);
    v8bf hi = *(const v8bf*)(p + 8);
    v16bf a;
#pragma unroll
    for (int i = 0; i < 8; ++i) { a[i] = lo[i]; a[i + 8] = hi[i]; }
    return a;
}

// B-matrix 32x16 (KxN): lane(0..15)=col N holds K=0..15, lanes 16..31 hold K=16..31.
// Weight slice staged in LDS as [gate][16 rows][32 k] bf16 -> one 32B ds read per lane.
__device__ __forceinline__ v16bf lds_b_frag(const bf16_t* sbuf, int g, int lane) {
    const bf16_t* p = sbuf + (g * 16 + (lane & 15)) * 32 + (lane >> 4) * 16;
    return *(const v16bf*)p;
}

// Cooperative fill: 256 threads x 16B = 4KB = 4 gates x (16 x 32) bf16 weight slice.
__device__ __forceinline__ void fill_b_tile(bf16_t* sbuf, const bf16_t* __restrict__ wt,
                                            int K, int H, int n0, int kk, int tid) {
    const int g  = tid >> 6;          // gate 0..3
    const int r  = (tid >> 2) & 15;   // row in 16-wide n tile
    const int cq = tid & 3;           // 8-element k chunk
    const bf16_t* src = wt + (long)(g * H + n0 + r) * K + kk + cq * 8;
    *(v8bf*)(sbuf + (g * 16 + r) * 32 + cq * 8) = *(const v8bf*)src;
}

__device__ __forceinline__ float sigmoidf_(float x) {
    return 1.0f / (1.0f + __expf(-x));
}

// ---------------- fused LSTM step -------------------------------------------
// g = [x_t, h_prev] @ [Wx; Wh] + bias ; gates ; c/h update.
// Block = 8 waves covering 8 batch-tiles of ONE hidden tile: the 4-gate weight
// slice is staged once per block per 32-K chunk (8x L2 traffic reduction),
// double-buffered. Chunk loop is FULLY UNROLLED (NC compile-time) so Wx/Wh
// selection, k offsets and buffer parity are constants -> no acc rotation movs.
template <int IN, int H>
__global__ __launch_bounds__(256) void lstm_step_kernel(
    const bf16_t* __restrict__ xA, long xA_stride,     // [B, IN]
    const bf16_t* __restrict__ WxT,                    // [4H, IN] transposed bf16
    const bf16_t* __restrict__ WhT,                    // [4H, H]  transposed bf16
    const float*  __restrict__ bias,                   // [4H] (bx+bh)
    const bf16_t* __restrict__ h_prev,                 // [B, H] bf16
    bf16_t*       __restrict__ h_next,                 // [B, H] bf16
    float*        __restrict__ c,                      // [B, H] fp32, in-place
    float*        __restrict__ h_f32,                  // [B, H] fp32 (last step survives)
    float*        __restrict__ h_seq,                  // [B, T, H] slot t, or nullptr
    int t)
{
    __shared__ __align__(16) bf16_t smem[2][4 * 16 * 32];   // 2 x 4KB weight slices

    const int tid    = threadIdx.x;
    const int lane   = tid & 31;
    const int wave   = tid >> 5;                 // 0..7
    const int MB     = (B_ / 16) / 8;            // batch tile groups per n-column (=2)
    const int m0     = ((blockIdx.x % MB) * 8 + wave) * 16;  // batch tile
    const int n0     = (blockIdx.x / MB) * 16;               // hidden tile

    constexpr int NCX = IN / 32;
    constexpr int NC  = IN / 32 + H / 32;        // = 40 for both layers

    v8f acc[4];
#pragma unroll
    for (int g = 0; g < 4; ++g)
#pragma unroll
        for (int r = 0; r < 8; ++r) acc[g][r] = 0.0f;

    // prime buffer 0 with chunk 0 (always an x-part chunk)
    fill_b_tile(smem[0], WxT, IN, H, n0, 0, tid);
    __syncthreads();

#pragma unroll
    for (int cc = 0; cc < NC; ++cc) {
        const bool xpart = (cc < NCX);                       // compile-time
        const int  kk    = xpart ? cc * 32 : (cc - NCX) * 32;

        // A fragment straight from global (tiny, L2-resident)
        v16bf a = xpart ? load_a_frag(xA,     xA_stride, m0, kk, lane)
                        : load_a_frag(h_prev, (long)H,   m0, kk, lane);

        // overlap: fill next chunk's weight slice into the other buffer
        if (cc + 1 < NC) {
            const bool nxp = (cc + 1 < NCX);
            const int  nkk = nxp ? (cc + 1) * 32 : (cc + 1 - NCX) * 32;
            fill_b_tile(smem[(cc + 1) & 1], nxp ? WxT : WhT, nxp ? IN : H, H,
                        n0, nkk, tid);
        }

        const bf16_t* sbuf = smem[cc & 1];
#pragma unroll
        for (int g = 0; g < 4; ++g) {
            v16bf b = lds_b_frag(sbuf, g, lane);
            acc[g] = __builtin_amdgcn_wmma_f32_16x16x32_bf16(
                false, a, false, b, (short)0, acc[g], false, false);
        }
        __syncthreads();
    }

    // ---- epilogue: C/D layout -> VGPR r, lanes 0-15: M=r ; lanes 16-31: M=r+8 ----
    const int n  = n0 + (lane & 15);
    const int mb = m0 + (lane >> 4) * 8;
    const float bi = bias[0 * H + n];
    const float bf = bias[1 * H + n];
    const float bo = bias[2 * H + n];
    const float bg = bias[3 * H + n];
#pragma unroll
    for (int r = 0; r < 8; ++r) {
        const int  m   = mb + r;
        const long idx = (long)m * H + n;
        float i_g = sigmoidf_(acc[0][r] + bi);
        float f_g = sigmoidf_(acc[1][r] + bf);
        float o_g = sigmoidf_(acc[2][r] + bo);
        float g_g = tanhf(acc[3][r] + bg);
        float cn  = f_g * c[idx] + i_g * g_g;
        float hn  = o_g * tanhf(cn);
        c[idx]      = cn;
        h_f32[idx]  = hn;
        h_next[idx] = (bf16_t)hn;
        if (h_seq) h_seq[(long)m * (T_ * H) + (long)t * H + n] = hn;
    }
}

// ---------------- preprocessing ----------------
__global__ void cvt_f32_bf16_kernel(const float* __restrict__ in,
                                    bf16_t* __restrict__ out, int n) {
    int i = blockIdx.x * blockDim.x + threadIdx.x;
    if (i < n) out[i] = (bf16_t)in[i];
}

// W: (4, K, H) fp32 row-major  ->  WT: (4H, K) bf16  (WT[g*H+h, k] = W[g,k,h])
__global__ void transpose_w_kernel(const float* __restrict__ W,
                                   bf16_t* __restrict__ WT, int K, int H) {
    long i = (long)blockIdx.x * blockDim.x + threadIdx.x;
    long total = 4l * K * H;
    if (i >= total) return;
    int k  = (int)(i % K);
    int gh = (int)(i / K);
    int g = gh / H, h = gh % H;
    WT[i] = (bf16_t)W[((long)g * K + k) * H + h];
}

__global__ void bias_combine_kernel(const float* __restrict__ bx,
                                    const float* __restrict__ bh,
                                    float* __restrict__ b, int n) {
    int i = blockIdx.x * blockDim.x + threadIdx.x;
    if (i < n) b[i] = bx[i] + bh[i];
}

__global__ void zero_words_kernel(unsigned int* __restrict__ p, long n) {
    long i = (long)blockIdx.x * blockDim.x + threadIdx.x;
    if (i < n) p[i] = 0u;
}

// ---------------- projection: flat (B,T,H1) reinterpreted as (B, 256, 128) rows ----
__global__ void projection_kernel(const float* __restrict__ h1_seq,
                                  const float* __restrict__ Wout,
                                  const float* __restrict__ bout,
                                  float* __restrict__ out) {
    int i = blockIdx.x * blockDim.x + threadIdx.x;          // over B*256*12
    if (i >= B_ * F_ * OUTL_) return;
    int o = i % OUTL_;
    int f = (i / OUTL_) % F_;
    int b = i / (OUTL_ * F_);
    const float* row = h1_seq + (long)b * (T_ * H1_) + (long)f * T_;
    float s = bout[o];
#pragma unroll 4
    for (int tt = 0; tt < T_; ++tt) s += row[tt] * Wout[tt * OUTL_ + o];
    out[i] = s;
}

// ---------------- h_n / c_n tail copy ----------------
__global__ void copy_tails_kernel(const float* __restrict__ h0l,
                                  const float* __restrict__ h1l,
                                  const float* __restrict__ c0,
                                  const float* __restrict__ c1,
                                  float* __restrict__ out) {
    int i = blockIdx.x * blockDim.x + threadIdx.x;
    const int S0 = B_ * H0_, S1 = B_ * H1_;
    if (i < S0)                       { out[i] = h0l[i]; return; }
    i -= S0;
    if (i < S1)                       { out[S0 + i] = h1l[i]; return; }
    i -= S1;
    if (i < S0)                       { out[S0 + S1 + i] = c0[i]; return; }
    i -= S0;
    if (i < S1)                       { out[S0 + S1 + S0 + i] = c1[i]; return; }
}

// ---------------- launcher ----------------
extern "C" void kernel_launch(void* const* d_in, const int* in_sizes, int n_in,
                              void* d_out, int out_size, void* d_ws, size_t ws_size,
                              hipStream_t stream) {
    const float* x    = (const float*)d_in[0];
    const float* Wx0  = (const float*)d_in[1];
    const float* bx0  = (const float*)d_in[2];
    const float* Wh0  = (const float*)d_in[3];
    const float* bh0  = (const float*)d_in[4];
    const float* Wx1  = (const float*)d_in[5];
    const float* bx1  = (const float*)d_in[6];
    const float* Wh1  = (const float*)d_in[7];
    const float* bh1  = (const float*)d_in[8];
    const float* Wout = (const float*)d_in[9];
    const float* bout = (const float*)d_in[10];
    float* out = (float*)d_out;

    // ---- carve workspace ----
    char* ws = (char*)d_ws;
    auto carve = [&](size_t bytes) { char* p = ws; ws += (bytes + 255) & ~(size_t)255; return p; };
    bf16_t* x_bf   = (bf16_t*)carve((size_t)B_ * T_ * F_ * 2);      // 16 MB
    bf16_t* Wx0T   = (bf16_t*)carve((size_t)4 * H0_ * F_ * 2);      //  2 MB
    bf16_t* Wh0T   = (bf16_t*)carve((size_t)4 * H0_ * H0_ * 2);     //  8 MB
    bf16_t* Wx1T   = (bf16_t*)carve((size_t)4 * H1_ * H0_ * 2);     //  2 MB
    bf16_t* Wh1T   = (bf16_t*)carve((size_t)4 * H1_ * H1_ * 2);     // 0.5 MB
    float*  bias0  = (float*)carve((size_t)4 * H0_ * 4);
    float*  bias1  = (float*)carve((size_t)4 * H1_ * 4);
    // zeroed region (contiguous): h0 ping-pong, h1 ping-pong, c0, c1
    char*   zbase  = ws;
    bf16_t* h0_bf  = (bf16_t*)carve((size_t)2 * B_ * H0_ * 2);
    bf16_t* h1_bf  = (bf16_t*)carve((size_t)2 * B_ * H1_ * 2);
    float*  c0     = (float*)carve((size_t)B_ * H0_ * 4);
    float*  c1     = (float*)carve((size_t)B_ * H1_ * 4);
    long    zwords = ((char*)ws - zbase) / 4;
    float*  h0_last = (float*)carve((size_t)B_ * H0_ * 4);
    float*  h1_last = (float*)carve((size_t)B_ * H1_ * 4);
    float*  h1_seq  = (float*)carve((size_t)B_ * T_ * H1_ * 4);     // 33.5 MB

    // ---- preprocessing ----
    {
        int n = B_ * T_ * F_;
        cvt_f32_bf16_kernel<<<(n + 255) / 256, 256, 0, stream>>>(x, x_bf, n);
    }
    transpose_w_kernel<<<(4 * F_ * H0_ + 255) / 256, 256, 0, stream>>>(Wx0, Wx0T, F_, H0_);
    transpose_w_kernel<<<(4 * H0_ * H0_ + 255) / 256, 256, 0, stream>>>(Wh0, Wh0T, H0_, H0_);
    transpose_w_kernel<<<(4 * H0_ * H1_ + 255) / 256, 256, 0, stream>>>(Wx1, Wx1T, H0_, H1_);
    transpose_w_kernel<<<(4 * H1_ * H1_ + 255) / 256, 256, 0, stream>>>(Wh1, Wh1T, H1_, H1_);
    bias_combine_kernel<<<(4 * H0_ + 255) / 256, 256, 0, stream>>>(bx0, bh0, bias0, 4 * H0_);
    bias_combine_kernel<<<(4 * H1_ + 255) / 256, 256, 0, stream>>>(bx1, bh1, bias1, 4 * H1_);
    zero_words_kernel<<<(int)((zwords + 255) / 256), 256, 0, stream>>>((unsigned int*)zbase, zwords);

    // ---- recurrence: 2 fused WMMA step-kernels per time step ----
    const size_t P0 = (size_t)B_ * H0_;   // h0 ping-pong stride (elements)
    const size_t P1 = (size_t)B_ * H1_;
    for (int t = 0; t < T_; ++t) {
        const int rp = t & 1, wp = (t + 1) & 1;
        // layer 0: 2 batch-groups x 64 hidden tiles = 128 blocks of 8 waves
        lstm_step_kernel<F_, H0_><<<2 * (H0_ / 16), 256, 0, stream>>>(
            x_bf + (long)t * F_, (long)T_ * F_,
            Wx0T, Wh0T, bias0,
            h0_bf + rp * P0, h0_bf + wp * P0, c0, h0_last, nullptr, t);
        // layer 1: consumes h0 just written; 2 x 16 = 32 blocks
        lstm_step_kernel<H0_, H1_><<<2 * (H1_ / 16), 256, 0, stream>>>(
            h0_bf + wp * P0, (long)H0_,
            Wx1T, Wh1T, bias1,
            h1_bf + rp * P1, h1_bf + wp * P1, c1, h1_last, h1_seq, t);
    }

    // ---- projection into d_out[0 : B*12*256)  (flat layout identical to reference) ----
    projection_kernel<<<(B_ * F_ * OUTL_ + 255) / 256, 256, 0, stream>>>(
        h1_seq, Wout, bout, out);

    // ---- h_n0, h_n1, c_n0, c_n1 ----
    {
        int tail = 2 * (B_ * H0_ + B_ * H1_);
        copy_tails_kernel<<<(tail + 255) / 256, 256, 0, stream>>>(
            h0_last, h1_last, c0, c1, out + (size_t)B_ * OUTL_ * F_);
    }
}
